// DecoderLayerWithContext_11905649345014
// MI455X (gfx1250) — compile-verified, matrix operand order checked by reading, not run on hardware
//
#include <hip/hip_runtime.h>

// ---------------------------------------------------------------------------
// Types
// ---------------------------------------------------------------------------
typedef __bf16 bf16_t;
typedef bf16_t bfvec16 __attribute__((ext_vector_type(16)));
typedef float  v8f     __attribute__((ext_vector_type(8)));
typedef unsigned short u16;
typedef unsigned int   u32;

union FragBF { uint4 u[2]; bfvec16 v; };

__device__ __forceinline__ u16 f2bf(float f) {
    u32 u = __float_as_uint(f);
    u32 r = u + 0x7FFFu + ((u >> 16) & 1u);   // round-to-nearest-even
    return (u16)(r >> 16);
}

__device__ __forceinline__ v8f wmma_bf16(bfvec16 a, bfvec16 b, v8f c) {
    return __builtin_amdgcn_wmma_f32_16x16x32_bf16(
        /*neg_a=*/false, a, /*neg_b=*/false, b,
        /*c_mod=*/(short)0, c, /*reuse_a=*/false, /*reuse_b=*/false);
}

// CDNA5 async global->LDS copy, 16B per lane (tracked by ASYNCcnt)
__device__ __forceinline__ void async_copy_b128(const void* gsrc, void* lds_dst) {
    unsigned long long ga = (unsigned long long)(uintptr_t)gsrc;
    unsigned           la = (unsigned)(uintptr_t)lds_dst;   // low 32 bits = LDS offset
    asm volatile("global_load_async_to_lds_b128 %0, %1, off"
                 :: "v"(la), "v"(ga) : "memory");
}
__device__ __forceinline__ void wait_async0() {
    asm volatile("s_wait_asynccnt 0x0" ::: "memory");
}

// ---------------------------------------------------------------------------
// fp32 -> bf16 convert (weights, context)
// ---------------------------------------------------------------------------
__global__ __launch_bounds__(256) void cvt_f32_bf16(const float* __restrict__ in,
                                                    u16* __restrict__ out, int n) {
    int i = blockIdx.x * blockDim.x + threadIdx.x;
    int stride = gridDim.x * blockDim.x;
    for (; i < n; i += stride) out[i] = f2bf(in[i]);
}

// ---------------------------------------------------------------------------
// LayerNorm over E=1024, one block (256 threads) per row, bf16 output
// ---------------------------------------------------------------------------
__device__ __forceinline__ float blockReduceSum256(float v) {
    __shared__ float sh[8];
    int lane = threadIdx.x & 31, wid = threadIdx.x >> 5;
    #pragma unroll
    for (int o = 16; o; o >>= 1) v += __shfl_xor(v, o);
    if (lane == 0) sh[wid] = v;
    __syncthreads();
    v = sh[0] + sh[1] + sh[2] + sh[3] + sh[4] + sh[5] + sh[6] + sh[7];
    __syncthreads();
    return v;
}

__global__ __launch_bounds__(256) void layernorm_bf16(const float* __restrict__ x,
                                                      const float* __restrict__ g,
                                                      const float* __restrict__ b,
                                                      u16* __restrict__ out) {
    const int E = 1024;
    int row = blockIdx.x, tid = threadIdx.x;
    const float4* xr = (const float4*)(x + (size_t)row * E);
    float4 v = xr[tid];
    float s = v.x + v.y + v.z + v.w;
    s = blockReduceSum256(s);
    float mu = s * (1.0f / E);
    float d0 = v.x - mu, d1 = v.y - mu, d2 = v.z - mu, d3 = v.w - mu;
    float q = d0 * d0 + d1 * d1 + d2 * d2 + d3 * d3;
    q = blockReduceSum256(q);
    float inv = rsqrtf(q * (1.0f / E) + 1e-5f);
    int c = tid * 4;
    float y0 = d0 * inv * g[c + 0] + b[c + 0];
    float y1 = d1 * inv * g[c + 1] + b[c + 1];
    float y2 = d2 * inv * g[c + 2] + b[c + 2];
    float y3 = d3 * inv * g[c + 3] + b[c + 3];
    uint2 pk;
    pk.x = (u32)f2bf(y0) | ((u32)f2bf(y1) << 16);
    pk.y = (u32)f2bf(y2) | ((u32)f2bf(y3) << 16);
    *(uint2*)(out + (size_t)row * E + c) = pk;
}

// ---------------------------------------------------------------------------
// bf16 WMMA GEMM: C[M,N] = A[M,K] @ W[K,N] (+bias)(+GELU)(+residual)
// 128x64 tile / 256 threads (8 waves), BK=32.
// Double-buffered LDS; A staged via async global->LDS, W transposed via regs.
// One s_wait_asynccnt + one barrier per K step.
// ---------------------------------------------------------------------------
template<bool BIAS, bool RES, bool GELU, bool OUTF, bool OUTB>
__global__ __launch_bounds__(256) void gemm_bf16_wmma(
    const u16* __restrict__ A, const u16* __restrict__ W,
    const float* __restrict__ bias, const float* __restrict__ residual,
    float* __restrict__ outF, u16* __restrict__ outB,
    int M, int N, int K)
{
    const int LDA = 40, LDB = 40;                         // padded LDS strides
    __shared__ __align__(16) u16 ldsA[2][128 * LDA];      // [row][k]
    __shared__ __align__(16) u16 ldsB[2][64 * LDB];       // [n][k] (transposed)

    int tid = threadIdx.x;
    int wid = tid >> 5, lane = tid & 31;
    int r16 = lane & 15, kh = lane >> 4;
    int wm = (wid & 3) * 32;     // wave row offset in tile
    int wn = (wid >> 2) * 32;    // wave col offset in tile
    int bm0 = blockIdx.x * 128;
    int bn0 = blockIdx.y * 64;

    // per-thread staging coordinates
    int arow0 = tid >> 2,        acol0 = (tid & 3) * 8;          // A chunk 0
    int arow1 = (tid + 256) >> 2, acol1 = ((tid + 256) & 3) * 8; // A chunk 1
    int wrow = tid >> 3,         wcol = (tid & 7) * 8;           // W chunk

    v8f acc[4];
    #pragma unroll
    for (int i = 0; i < 4; ++i)
        acc[i] = (v8f){0.f, 0.f, 0.f, 0.f, 0.f, 0.f, 0.f, 0.f};

    int nk = K >> 5;

    // ---- prologue: stage k-tile 0 into buffer 0 ----
    async_copy_b128(A + (size_t)(bm0 + arow0) * K + acol0, &ldsA[0][arow0 * LDA + acol0]);
    async_copy_b128(A + (size_t)(bm0 + arow1) * K + acol1, &ldsA[0][arow1 * LDA + acol1]);
    {
        uint4 ld = *(const uint4*)(W + (size_t)wrow * N + bn0 + wcol);
        u32 wds[4] = {ld.x, ld.y, ld.z, ld.w};
        #pragma unroll
        for (int j = 0; j < 4; ++j) {
            ldsB[0][(wcol + 2 * j + 0) * LDB + wrow] = (u16)(wds[j] & 0xffff);
            ldsB[0][(wcol + 2 * j + 1) * LDB + wrow] = (u16)(wds[j] >> 16);
        }
    }

    for (int kt = 0; kt < nk; ++kt) {
        int p = kt & 1;
        wait_async0();         // my async writes into buf p are done
        __syncthreads();       // everyone's stage-p writes visible, stage-p^1 reads retired

        // issue stage kt+1 into buffer p^1 (overlaps with WMMA below)
        uint4 bNext;
        bool more = (kt + 1) < nk;
        if (more) {
            int kb = (kt + 1) * 32;
            async_copy_b128(A + (size_t)(bm0 + arow0) * K + kb + acol0,
                            &ldsA[p ^ 1][arow0 * LDA + acol0]);
            async_copy_b128(A + (size_t)(bm0 + arow1) * K + kb + acol1,
                            &ldsA[p ^ 1][arow1 * LDA + acol1]);
            bNext = *(const uint4*)(W + (size_t)(kb + wrow) * N + bn0 + wcol);
        }

        // compute on buffer p
        FragBF aF[2];
        #pragma unroll
        for (int fm = 0; fm < 2; ++fm) {
            int base = (wm + fm * 16 + r16) * LDA + kh * 8;
            aF[fm].u[0] = *(const uint4*)&ldsA[p][base];
            aF[fm].u[1] = *(const uint4*)&ldsA[p][base + 16];
        }
        #pragma unroll
        for (int fn = 0; fn < 2; ++fn) {
            FragBF bF;
            int base = (wn + fn * 16 + r16) * LDB + kh * 8;
            bF.u[0] = *(const uint4*)&ldsB[p][base];
            bF.u[1] = *(const uint4*)&ldsB[p][base + 16];
            acc[0 * 2 + fn] = wmma_bf16(aF[0].v, bF.v, acc[0 * 2 + fn]);
            acc[1 * 2 + fn] = wmma_bf16(aF[1].v, bF.v, acc[1 * 2 + fn]);
        }

        // transpose-store next W tile after compute is issued
        if (more) {
            u32 wds[4] = {bNext.x, bNext.y, bNext.z, bNext.w};
            #pragma unroll
            for (int j = 0; j < 4; ++j) {
                ldsB[p ^ 1][(wcol + 2 * j + 0) * LDB + wrow] = (u16)(wds[j] & 0xffff);
                ldsB[p ^ 1][(wcol + 2 * j + 1) * LDB + wrow] = (u16)(wds[j] >> 16);
            }
        }
    }

    // epilogue
    #pragma unroll
    for (int fm = 0; fm < 2; ++fm) {
        #pragma unroll
        for (int fn = 0; fn < 2; ++fn) {
            #pragma unroll
            for (int r = 0; r < 8; ++r) {
                int grow = bm0 + wm + fm * 16 + r + 8 * kh;
                int gcol = bn0 + wn + fn * 16 + r16;
                float v = acc[fm * 2 + fn][r];
                if (BIAS) v += bias[gcol];
                if (GELU) v = 0.5f * v * (1.0f + erff(v * 0.70710678118654752f));
                if (RES)  v += residual[(size_t)grow * N + gcol];
                if (OUTF) outF[(size_t)grow * N + gcol] = v;
                if (OUTB) outB[(size_t)grow * N + gcol] = f2bf(v);
            }
        }
    }
}

// ---------------------------------------------------------------------------
// Flash attention (bf16 q/k/v [B,T,H,64], bf16 out), online softmax.
// Block = 128 threads (4 waves); each wave owns 16 q rows; S chunk = 64.
// K tile staged via async global->LDS; V transposed via regs.
// ---------------------------------------------------------------------------
template<bool CAUSAL>
__global__ __launch_bounds__(128) void flash_attn_bf16(
    const u16* __restrict__ q, const u16* __restrict__ k, const u16* __restrict__ v,
    u16* __restrict__ o, int T, int Tk, int H)
{
    const int D = 64, LDK = 72;
    const float SCALE = 0.03125f;                 // 1/sqrt(E=1024)
    __shared__ __align__(16) u16 ldsK[64 * LDK];  // [s][d]
    __shared__ __align__(16) u16 ldsV[64 * LDK];  // [d][s] transposed
    __shared__ __align__(16) u16 ldsP[4 * 16 * LDK]; // per-wave P staging [q][s]

    int tid = threadIdx.x, wid = tid >> 5, lane = tid & 31;
    int r16 = lane & 15, kh = lane >> 4;
    int qTile = blockIdx.x, h = blockIdx.y, b = blockIdx.z;
    int q0 = qTile * 64;

    // load this wave's Q fragments (16 rows x 64 d)
    FragBF qA[2];
    {
        int qrow = q0 + wid * 16 + r16;
        const u16* qp = q + ((size_t)(b * T + qrow) * H + h) * D;
        #pragma unroll
        for (int ks = 0; ks < 2; ++ks) {
            qA[ks].u[0] = *(const uint4*)(qp + ks * 32 + kh * 8);
            qA[ks].u[1] = *(const uint4*)(qp + ks * 32 + 16 + kh * 8);
        }
    }

    v8f O[4];
    float m[8], l[8];
    #pragma unroll
    for (int n = 0; n < 4; ++n)
        O[n] = (v8f){0.f, 0.f, 0.f, 0.f, 0.f, 0.f, 0.f, 0.f};
    #pragma unroll
    for (int r = 0; r < 8; ++r) { m[r] = -1e30f; l[r] = 0.f; }

    int nS = CAUSAL ? (qTile + 1) : (Tk >> 6);
    for (int st = 0; st < nS; ++st) {
        int s0 = st * 64;
        bool diag = CAUSAL && (st == qTile);      // only chunk needing a mask
        __syncthreads();
        // cooperative K/V staging: K async straight-copy, V transposed via regs
        #pragma unroll
        for (int i = 0; i < 4; ++i) {
            int chunk = tid + i * 128;
            int srow = chunk >> 3, dc = (chunk & 7) * 8;
            size_t gbase = ((size_t)(b * Tk + s0 + srow) * H + h) * D + dc;
            async_copy_b128(k + gbase, &ldsK[srow * LDK + dc]);
            uint4 vv = *(const uint4*)(v + gbase);
            u32 wd[4] = {vv.x, vv.y, vv.z, vv.w};
            #pragma unroll
            for (int j = 0; j < 4; ++j) {
                ldsV[(dc + 2 * j + 0) * LDK + srow] = (u16)(wd[j] & 0xffff);
                ldsV[(dc + 2 * j + 1) * LDK + srow] = (u16)(wd[j] >> 16);
            }
        }
        wait_async0();
        __syncthreads();

        // scores: 16q x 64s via WMMA (K-dim = D = 2 x 32)
        v8f sc[4];
        #pragma unroll
        for (int n = 0; n < 4; ++n) {
            FragBF kb0, kb1;
            int cb = (n * 16 + r16) * LDK;
            kb0.u[0] = *(const uint4*)&ldsK[cb + kh * 8];
            kb0.u[1] = *(const uint4*)&ldsK[cb + 16 + kh * 8];
            kb1.u[0] = *(const uint4*)&ldsK[cb + 32 + kh * 8];
            kb1.u[1] = *(const uint4*)&ldsK[cb + 48 + kh * 8];
            v8f z = (v8f){0.f, 0.f, 0.f, 0.f, 0.f, 0.f, 0.f, 0.f};
            v8f t = wmma_bf16(qA[0].v, kb0.v, z);
            sc[n]  = wmma_bf16(qA[1].v, kb1.v, t);
        }

        // scale + (diagonal-only) causal mask + row max
        float mch[8];
        #pragma unroll
        for (int r = 0; r < 8; ++r) mch[r] = -3e30f;
        #pragma unroll
        for (int n = 0; n < 4; ++n) {
            #pragma unroll
            for (int r = 0; r < 8; ++r) {
                float val = sc[n][r] * SCALE;
                if (diag) {
                    int sg = s0 + n * 16 + r16;
                    int qg = q0 + wid * 16 + r + 8 * kh;
                    if (sg > qg) val = -3e30f;
                }
                sc[n][r] = val;
                mch[r] = fmaxf(mch[r], val);
            }
        }
        #pragma unroll
        for (int r = 0; r < 8; ++r) {
            #pragma unroll
            for (int off = 8; off; off >>= 1)
                mch[r] = fmaxf(mch[r], __shfl_xor(mch[r], off));
        }
        float alpha[8];
        #pragma unroll
        for (int r = 0; r < 8; ++r) {
            float mn = fmaxf(m[r], mch[r]);
            alpha[r] = __expf(m[r] - mn);
            m[r] = mn;
        }

        // P = exp(sc - m); stage to per-wave LDS in A-fragment-friendly layout
        float rs[8];
        #pragma unroll
        for (int r = 0; r < 8; ++r) rs[r] = 0.f;
        #pragma unroll
        for (int n = 0; n < 4; ++n) {
            #pragma unroll
            for (int r = 0; r < 8; ++r) {
                float p = __expf(sc[n][r] - m[r]);
                rs[r] += p;
                ldsP[wid * 16 * LDK + (r + 8 * kh) * LDK + n * 16 + r16] = f2bf(p);
            }
        }
        #pragma unroll
        for (int r = 0; r < 8; ++r) {
            #pragma unroll
            for (int off = 8; off; off >>= 1)
                rs[r] += __shfl_xor(rs[r], off);
            l[r] = l[r] * alpha[r] + rs[r];
        }

        // cross-lane LDS staging: force store->load ordering within the wave
        asm volatile("s_wait_dscnt 0x0" ::: "memory");

        FragBF pA0, pA1;
        {
            int pb = wid * 16 * LDK + r16 * LDK;
            pA0.u[0] = *(const uint4*)&ldsP[pb + kh * 8];
            pA0.u[1] = *(const uint4*)&ldsP[pb + 16 + kh * 8];
            pA1.u[0] = *(const uint4*)&ldsP[pb + 32 + kh * 8];
            pA1.u[1] = *(const uint4*)&ldsP[pb + 48 + kh * 8];
        }

        // O = diag(alpha) * O + P @ V
        #pragma unroll
        for (int n = 0; n < 4; ++n) {
            #pragma unroll
            for (int r = 0; r < 8; ++r) O[n][r] *= alpha[r];
            FragBF vb0, vb1;
            int db = (n * 16 + r16) * LDK;
            vb0.u[0] = *(const uint4*)&ldsV[db + kh * 8];
            vb0.u[1] = *(const uint4*)&ldsV[db + 16 + kh * 8];
            vb1.u[0] = *(const uint4*)&ldsV[db + 32 + kh * 8];
            vb1.u[1] = *(const uint4*)&ldsV[db + 48 + kh * 8];
            O[n] = wmma_bf16(pA0.v, vb0.v, O[n]);
            O[n] = wmma_bf16(pA1.v, vb1.v, O[n]);
        }
    }

    // normalize + store bf16
    #pragma unroll
    for (int r = 0; r < 8; ++r) {
        int qg = q0 + wid * 16 + r + 8 * kh;
        u16* op = o + ((size_t)(b * T + qg) * H + h) * D;
        float invl = 1.0f / l[r];
        #pragma unroll
        for (int n = 0; n < 4; ++n)
            op[n * 16 + r16] = f2bf(O[n][r] * invl);
    }
}

// ---------------------------------------------------------------------------
// Orchestration
// ---------------------------------------------------------------------------
extern "C" void kernel_launch(void* const* d_in, const int* in_sizes, int n_in,
                              void* d_out, int out_size, void* d_ws, size_t ws_size,
                              hipStream_t stream) {
    (void)in_sizes; (void)n_in; (void)out_size; (void)ws_size;
    const int E = 1024, H = 16, T = 2048, S = 2048, Bb = 2;
    const int M = Bb * T;               // 4096 rows
    const int EE = E * E;               // 1,048,576
    const int E4 = E * 4 * E;           // 4,194,304

    const float* x      = (const float*)d_in[0];
    const float* ctx    = (const float*)d_in[1];
    const float* ln1_g  = (const float*)d_in[2];
    const float* ln1_b  = (const float*)d_in[3];
    const float* ln2_g  = (const float*)d_in[4];
    const float* ln2_b  = (const float*)d_in[5];
    const float* ln3_g  = (const float*)d_in[6];
    const float* ln3_b  = (const float*)d_in[7];
    const float* Wq_s   = (const float*)d_in[8];
    const float* Wk_s   = (const float*)d_in[9];
    const float* Wv_s   = (const float*)d_in[10];
    const float* proj_s = (const float*)d_in[11];
    const float* pbs    = (const float*)d_in[12];
    const float* Wq_c   = (const float*)d_in[13];
    const float* Wk_c   = (const float*)d_in[14];
    const float* Wv_c   = (const float*)d_in[15];
    const float* proj_c = (const float*)d_in[16];
    const float* pbc    = (const float*)d_in[17];
    const float* w1     = (const float*)d_in[18];
    const float* b1     = (const float*)d_in[19];
    const float* w2     = (const float*)d_in[20];
    const float* b2     = (const float*)d_in[21];
    float* out = (float*)d_out;

    // workspace bump allocator (256B aligned)
    char* ws = (char*)d_ws;
    size_t off = 0;
    auto alloc = [&](size_t bytes) -> char* {
        char* p = ws + off;
        off += (bytes + 255) & ~(size_t)255;
        return p;
    };
    u16* wb[10];
    const float* wsrc[10] = {Wq_s, Wk_s, Wv_s, proj_s, Wq_c, Wk_c, Wv_c, proj_c, w1, w2};
    const size_t wnel[10] = {(size_t)EE, EE, EE, EE, EE, EE, EE, EE, (size_t)E4, (size_t)E4};
    for (int i = 0; i < 10; ++i) wb[i] = (u16*)alloc(wnel[i] * 2);
    u16*   actA  = (u16*)alloc((size_t)M * E * 2);
    u16*   qb    = (u16*)alloc((size_t)M * E * 2);
    u16*   kb    = (u16*)alloc((size_t)M * E * 2);
    u16*   vb    = (u16*)alloc((size_t)M * E * 2);
    u16*   attnb = (u16*)alloc((size_t)M * E * 2);
    float* x2    = (float*)alloc((size_t)M * E * 4);
    u16*   ff1   = (u16*)alloc((size_t)M * 4 * E * 2);

    // 0) weights -> bf16
    for (int i = 0; i < 10; ++i)
        cvt_f32_bf16<<<1024, 256, 0, stream>>>(wsrc[i], wb[i], (int)wnel[i]);

    dim3 gN1(M / 128, E / 64);      // N = 1024
    dim3 gN4(M / 128, 4 * E / 64);  // N = 4096
    dim3 gFA(T / 64, H, Bb);

    // ---- self-attention sublayer ----
    layernorm_bf16<<<M, 256, 0, stream>>>(x, ln1_g, ln1_b, actA);
    gemm_bf16_wmma<false,false,false,false,true><<<gN1, 256, 0, stream>>>(actA, wb[0], nullptr, nullptr, nullptr, qb, M, E, E);
    gemm_bf16_wmma<false,false,false,false,true><<<gN1, 256, 0, stream>>>(actA, wb[1], nullptr, nullptr, nullptr, kb, M, E, E);
    gemm_bf16_wmma<false,false,false,false,true><<<gN1, 256, 0, stream>>>(actA, wb[2], nullptr, nullptr, nullptr, vb, M, E, E);
    flash_attn_bf16<true><<<gFA, 128, 0, stream>>>(qb, kb, vb, attnb, T, T, H);
    gemm_bf16_wmma<true,true,false,true,false><<<gN1, 256, 0, stream>>>(attnb, wb[3], pbs, x, x2, nullptr, M, E, E);

    // ---- cross-attention sublayer (context not normalized) ----
    layernorm_bf16<<<M, 256, 0, stream>>>(x2, ln2_g, ln2_b, actA);
    gemm_bf16_wmma<false,false,false,false,true><<<gN1, 256, 0, stream>>>(actA, wb[4], nullptr, nullptr, nullptr, qb, M, E, E);
    cvt_f32_bf16<<<1024, 256, 0, stream>>>(ctx, attnb, Bb * S * E);   // reuse attnb as bf16 context
    gemm_bf16_wmma<false,false,false,false,true><<<gN1, 256, 0, stream>>>(attnb, wb[5], nullptr, nullptr, nullptr, kb, Bb * S, E, E);
    gemm_bf16_wmma<false,false,false,false,true><<<gN1, 256, 0, stream>>>(attnb, wb[6], nullptr, nullptr, nullptr, vb, Bb * S, E, E);
    flash_attn_bf16<false><<<gFA, 128, 0, stream>>>(qb, kb, vb, actA, T, S, H);
    gemm_bf16_wmma<true,true,false,true,false><<<gN1, 256, 0, stream>>>(actA, wb[7], pbc, x2, out, nullptr, M, E, E);

    // ---- FFN sublayer (exact-erf GELU) ----
    layernorm_bf16<<<M, 256, 0, stream>>>(out, ln3_g, ln3_b, attnb);
    gemm_bf16_wmma<true,false,true,false,true><<<gN4, 256, 0, stream>>>(attnb, wb[8], b1, nullptr, nullptr, ff1, M, 4 * E, E);
    gemm_bf16_wmma<true,true,false,true,false><<<gN1, 256, 0, stream>>>(ff1, wb[9], b2, out, out, nullptr, M, E, 4 * E);
}
